// MultiHeadedAttention_39221641347173
// MI455X (gfx1250) — matrix operand order repaired and checked by reading, not compiled, not run on hardware
//
#include <hip/hip_runtime.h>

typedef __attribute__((ext_vector_type(16))) __bf16 v16bf;
typedef __attribute__((ext_vector_type(8)))  float  floatx8;

#define DMODEL 512
#define SEQLEN 2048
#define LSTR   72   // LDS row stride (halfwords): 36-dword bank rotation, 16B-aligned rows

__device__ __forceinline__ unsigned short f2bfu(float f) {
    union { __bf16 h; unsigned short u; } c; c.h = (__bf16)f; return c.u;
}
__device__ __forceinline__ unsigned pkbf2(float a, float b) {
    union { __bf16 h[2]; unsigned u; } c;
    c.h[0] = (__bf16)a; c.h[1] = (__bf16)b; return c.u;
}

// gfx1250 async global->LDS DMA (ASYNCcnt-tracked), 16B per lane.
__device__ __forceinline__ void async_ld_b128(unsigned lds_addr, const void* gaddr) {
    asm volatile("global_load_async_to_lds_b128 %0, %1, off"
                 :: "v"(lds_addr), "v"(gaddr) : "memory");
}
__device__ __forceinline__ void wait_async0() {
    asm volatile("s_wait_asynccnt 0" ::: "memory");
}
__device__ __forceinline__ unsigned lds_off(const void* p) {
    return (unsigned)(unsigned long long)p;   // low 32 bits of generic LDS addr = LDS offset
}

// A fragment (16x32 bf16): lane L -> row L&15; hi half selects K banks.
__device__ __forceinline__ void frag_ld_a(const unsigned* p, int hi, unsigned* f) {
#pragma unroll
    for (int j = 0; j < 4; ++j) f[j]     = p[hi * 4 + j];
#pragma unroll
    for (int j = 0; j < 4; ++j) f[4 + j] = p[8 + hi * 4 + j];
}
// B fragment (32x16 bf16) from [N][K]-transposed storage: lane L -> col L&15.
__device__ __forceinline__ void frag_ld_b(const unsigned* p, int hi, unsigned* f) {
#pragma unroll
    for (int j = 0; j < 4; ++j) f[j]     = p[hi * 4 + j];
#pragma unroll
    for (int j = 0; j < 4; ++j) f[4 + j] = p[8 + hi * 4 + j];
}

__device__ __forceinline__ floatx8 wmma_bf16(const unsigned* a, const unsigned* b, floatx8 c) {
    union U { unsigned u[8]; v16bf v; } A, B;
#pragma unroll
    for (int i = 0; i < 8; ++i) { A.u[i] = a[i]; B.u[i] = b[i]; }
    return __builtin_amdgcn_wmma_f32_16x16x32_bf16(false, A.v, false, B.v,
                                                   (short)0, c, false, false);
}

// out[c,n] = sum_i W[c,i]*X[b,i,n] + bias[c].  Block: 128c x 64n, wave: 2x2 tiles.
// Software-pipelined k-loop with double-buffered LDS (1 barrier / k-step).
// MODE 0: bf16 -> outB[b][h=c&7][n][dd=c>>3];  MODE 1: f32 NT -> outF[b][c][n].
template <int MODE>
__global__ __launch_bounds__(256) void gemm_proj(
    const float* __restrict__ W, const float* __restrict__ X,
    const float* __restrict__ bias, unsigned short* __restrict__ outB,
    float* __restrict__ outF)
{
    __shared__ unsigned short As[2][128 * LSTR];  // [c][k]
    __shared__ unsigned short Bs[2][64 * LSTR];   // [n][k] (transposed)

    const int n0 = blockIdx.x * 64, c0 = blockIdx.y * 128, b = blockIdx.z;
    const float* Xb = X + (size_t)b * DMODEL * SEQLEN;
    const int tid = threadIdx.x, lane = tid & 31, wave = tid >> 5;
    const int hi = lane >> 4, lo = lane & 15;
    const int ct = (wave & 3) * 2, nt = (wave >> 2) * 2;

    const int rA = tid >> 1, khA = (tid & 1) * 16;   // A stage: 16 k-elems of one c-row
    const int kkB = tid >> 3, noB = (tid & 7) * 8;   // B stage: 8 n-elems of one k-row
    const float* Arow = W + (size_t)(c0 + rA) * DMODEL + khA;
    const float* Brow = Xb + (size_t)kkB * SEQLEN + n0 + noB;

    float4 a[4], xv[2];
    {   // prologue: stage k=0 and park in buffer 0
        const float4* s4 = (const float4*)Arow;
        a[0] = s4[0]; a[1] = s4[1]; a[2] = s4[2]; a[3] = s4[3];
        const float4* b4 = (const float4*)Brow;
        xv[0] = b4[0]; xv[1] = b4[1];
        unsigned* d32 = (unsigned*)&As[0][rA * LSTR + khA];
        d32[0] = pkbf2(a[0].x, a[0].y); d32[1] = pkbf2(a[0].z, a[0].w);
        d32[2] = pkbf2(a[1].x, a[1].y); d32[3] = pkbf2(a[1].z, a[1].w);
        d32[4] = pkbf2(a[2].x, a[2].y); d32[5] = pkbf2(a[2].z, a[2].w);
        d32[6] = pkbf2(a[3].x, a[3].y); d32[7] = pkbf2(a[3].z, a[3].w);
        float v[8] = {xv[0].x, xv[0].y, xv[0].z, xv[0].w, xv[1].x, xv[1].y, xv[1].z, xv[1].w};
#pragma unroll
        for (int i = 0; i < 8; ++i) Bs[0][(noB + i) * LSTR + kkB] = f2bfu(v[i]);
    }

    floatx8 acc[2][2] = {};
    const int NK = DMODEL / 32;
    for (int i = 0; i < NK; ++i) {
        const int cur = i & 1, nxt = cur ^ 1;
        __syncthreads();                               // buf[cur] visible; buf[nxt] free
        const bool hn = (i + 1 < NK);
        if (hn) {                                      // prefetch next k-slice into regs
            const float4* s4 = (const float4*)(Arow + (i + 1) * 32);
            a[0] = s4[0]; a[1] = s4[1]; a[2] = s4[2]; a[3] = s4[3];
            const float4* b4 = (const float4*)(Brow + (size_t)(i + 1) * 32 * SEQLEN);
            xv[0] = b4[0]; xv[1] = b4[1];
        }
        unsigned af[2][8], bfr[2][8];
#pragma unroll
        for (int u = 0; u < 2; ++u)
            frag_ld_a((const unsigned*)&As[cur][((ct + u) * 16 + lo) * LSTR], hi, af[u]);
#pragma unroll
        for (int j = 0; j < 2; ++j)
            frag_ld_b((const unsigned*)&Bs[cur][((nt + j) * 16 + lo) * LSTR], hi, bfr[j]);
#pragma unroll
        for (int u = 0; u < 2; ++u)
#pragma unroll
            for (int j = 0; j < 2; ++j)
                acc[u][j] = wmma_bf16(af[u], bfr[j], acc[u][j]);
        if (hn) {                                      // convert + park into buf[nxt]
            unsigned* d32 = (unsigned*)&As[nxt][rA * LSTR + khA];
            d32[0] = pkbf2(a[0].x, a[0].y); d32[1] = pkbf2(a[0].z, a[0].w);
            d32[2] = pkbf2(a[1].x, a[1].y); d32[3] = pkbf2(a[1].z, a[1].w);
            d32[4] = pkbf2(a[2].x, a[2].y); d32[5] = pkbf2(a[2].z, a[2].w);
            d32[6] = pkbf2(a[3].x, a[3].y); d32[7] = pkbf2(a[3].z, a[3].w);
            float v[8] = {xv[0].x, xv[0].y, xv[0].z, xv[0].w, xv[1].x, xv[1].y, xv[1].z, xv[1].w};
#pragma unroll
            for (int q = 0; q < 8; ++q) Bs[nxt][(noB + q) * LSTR + kkB] = f2bfu(v[q]);
        }
    }

    const int nbase = n0 + nt * 16 + lo;
#pragma unroll
    for (int u = 0; u < 2; ++u) {
#pragma unroll
        for (int r = 0; r < 8; ++r) {
            int c = c0 + (ct + u) * 16 + r + 8 * hi;
            float bv = bias[c];
            float v0 = acc[u][0][r] + bv, v1 = acc[u][1][r] + bv;
            if (MODE == 0) {
                int h = c & 7, dd = c >> 3;
                unsigned short* p = outB + (((size_t)(b * 8 + h) * SEQLEN + nbase) << 6) + dd;
                p[0]       = f2bfu(v0);
                p[16 * 64] = f2bfu(v1);
            } else {
                float* p = outF + ((size_t)b * DMODEL + c) * SEQLEN + nbase;
                __builtin_nontemporal_store(v0, p);
                __builtin_nontemporal_store(v1, p + 16);
            }
        }
    }
}

// Flash attention: block = 8 waves, 128 query rows, one (b,h).
// K tiles via async DMA, V via staged transpose; double-buffered, pipelined.
__global__ __launch_bounds__(256) void attn(
    const unsigned short* __restrict__ Qb, const unsigned short* __restrict__ Kb,
    const unsigned short* __restrict__ Vb, const int* __restrict__ Mask,
    float* __restrict__ Xout)
{
    __shared__ unsigned short Qs[128 * LSTR];       // [n][dd]
    __shared__ unsigned short Ks[2][64 * LSTR];     // [m][dd]
    __shared__ unsigned short Vs[2][64 * LSTR];     // [dd][m] (transposed)
    __shared__ unsigned short Ps[8][16 * LSTR];     // per-wave P staging

    const int h = blockIdx.y, b = blockIdx.z;
    const int n0 = blockIdx.x * 128;
    const int tid = threadIdx.x, lane = tid & 31, wave = tid >> 5;
    const int hi = lane >> 4, lo = lane & 15;
    const unsigned short* Qh = Qb + ((size_t)(b * 8 + h) << 17);   // *SEQLEN*64
    const unsigned short* Kh = Kb + ((size_t)(b * 8 + h) << 17);
    const unsigned short* Vh = Vb + ((size_t)(b * 8 + h) << 17);

    const int vm = tid >> 2, vd0 = (tid & 3) * 16;   // V staging slot
    union { uint4 q[2]; unsigned short s[16]; } vst;

    {   // prologue: Q tile + K tile 0 via async DMA; V tile 0 staged
        for (int i = tid; i < 128 * 8; i += 256) {
            int r = i >> 3, c = i & 7;
            async_ld_b128(lds_off(&Qs[r * LSTR + c * 8]),
                          Qh + (size_t)(n0 + r) * 64 + c * 8);
        }
        for (int i = tid; i < 64 * 8; i += 256) {
            int r = i >> 3, c = i & 7;
            async_ld_b128(lds_off(&Ks[0][r * LSTR + c * 8]), Kh + (size_t)r * 64 + c * 8);
        }
        const uint4* vsrc = (const uint4*)(Vh + (size_t)vm * 64 + vd0);
        vst.q[0] = vsrc[0]; vst.q[1] = vsrc[1];
        wait_async0();
#pragma unroll
        for (int j = 0; j < 16; ++j) Vs[0][(vd0 + j) * LSTR + vm] = vst.s[j];
        __syncthreads();
    }

    unsigned qf[2][8];
    frag_ld_a((const unsigned*)&Qs[(wave * 16 + lo) * LSTR] + 0,  hi, qf[0]);
    frag_ld_a((const unsigned*)&Qs[(wave * 16 + lo) * LSTR] + 16, hi, qf[1]);

    float mrow[8], lrow[8];
    floatx8 Of[4] = {};
#pragma unroll
    for (int r = 0; r < 8; ++r) { mrow[r] = -__builtin_inff(); lrow[r] = 0.f; }

    const int nrow0 = n0 + wave * 16;
    const int* mlane = Mask + ((size_t)b * SEQLEN + nrow0 + hi * 8) * SEQLEN + lo;

    const int NT = SEQLEN / 64;
    for (int it = 0; it < NT; ++it) {
        const int cur = it & 1, nxt = cur ^ 1;
        const bool hn = (it + 1 < NT);
        const int m0 = it * 64;
        if (hn) {                                    // prefetch tile it+1
            for (int i = tid; i < 64 * 8; i += 256) {
                int r = i >> 3, c = i & 7;
                async_ld_b128(lds_off(&Ks[nxt][r * LSTR + c * 8]),
                              Kh + (size_t)(m0 + 64 + r) * 64 + c * 8);
            }
            const uint4* vsrc = (const uint4*)(Vh + (size_t)(m0 + 64 + vm) * 64 + vd0);
            vst.q[0] = vsrc[0]; vst.q[1] = vsrc[1];
        }

        floatx8 S[4];                                // S = Q K^T on tile `cur`
#pragma unroll
        for (int t = 0; t < 4; ++t) {
            floatx8 s = {};
#pragma unroll
            for (int kc = 0; kc < 2; ++kc) {
                unsigned bfr[8];
                frag_ld_b((const unsigned*)&Ks[cur][(t * 16 + lo) * LSTR] + kc * 16, hi, bfr);
                s = wmma_bf16(qf[kc], bfr, s);
            }
            S[t] = s;
        }
#pragma unroll
        for (int t = 0; t < 4; ++t) {                // scale + mask (streamed NT)
#pragma unroll
            for (int r = 0; r < 8; ++r) {
                int mv = __builtin_nontemporal_load(mlane + (size_t)r * SEQLEN + m0 + t * 16);
                float sv = S[t][r] * 0.125f;
                S[t][r] = mv ? sv : -3.38953139e38f;
            }
        }
#pragma unroll
        for (int r = 0; r < 8; ++r) {                // online softmax per row
            float mx = fmaxf(fmaxf(S[0][r], S[1][r]), fmaxf(S[2][r], S[3][r]));
            mx = fmaxf(mx, __shfl_xor(mx, 1, 32));
            mx = fmaxf(mx, __shfl_xor(mx, 2, 32));
            mx = fmaxf(mx, __shfl_xor(mx, 4, 32));
            mx = fmaxf(mx, __shfl_xor(mx, 8, 32));
            float mnew = fmaxf(mrow[r], mx);
            float corr = __expf(mrow[r] - mnew);
            float p[4], rs = 0.f;
#pragma unroll
            for (int t = 0; t < 4; ++t) { p[t] = __expf(S[t][r] - mnew); rs += p[t]; }
            rs += __shfl_xor(rs, 1, 32);
            rs += __shfl_xor(rs, 2, 32);
            rs += __shfl_xor(rs, 4, 32);
            rs += __shfl_xor(rs, 8, 32);
            lrow[r] = lrow[r] * corr + rs;
            mrow[r] = mnew;
#pragma unroll
            for (int t2 = 0; t2 < 4; ++t2) Of[t2][r] = Of[t2][r] * corr;
#pragma unroll
            for (int t = 0; t < 4; ++t)
                Ps[wave][(r + 8 * hi) * LSTR + t * 16 + lo] = f2bfu(p[t]);
        }
#pragma unroll
        for (int kc = 0; kc < 2; ++kc) {             // O += P V on tile `cur`
            unsigned pf[8];
            frag_ld_a((const unsigned*)&Ps[wave][lo * LSTR] + kc * 16, hi, pf);
#pragma unroll
            for (int t2 = 0; t2 < 4; ++t2) {
                unsigned vf[8];
                frag_ld_b((const unsigned*)&Vs[cur][(t2 * 16 + lo) * LSTR] + kc * 16, hi, vf);
                Of[t2] = wmma_bf16(pf, vf, Of[t2]);
            }
        }
        if (hn) {                                    // park V(it+1), close the pipe stage
#pragma unroll
            for (int j = 0; j < 16; ++j) Vs[nxt][(vd0 + j) * LSTR + vm] = vst.s[j];
            wait_async0();
            __syncthreads();
        }
    }
#pragma unroll
    for (int r = 0; r < 8; ++r) {                    // normalize + store fp32
        float inv = 1.0f / lrow[r];
        int n = n0 + wave * 16 + r + 8 * hi;
#pragma unroll
        for (int t2 = 0; t2 < 4; ++t2) {
            int c = (t2 * 16 + lo) * 8 + h;
            Xout[((size_t)b * DMODEL + c) * SEQLEN + n] = Of[t2][r] * inv;
        }
    }
}

extern "C" void kernel_launch(void* const* d_in, const int* in_sizes, int n_in,
                              void* d_out, int out_size, void* d_ws, size_t ws_size,
                              hipStream_t stream) {
    const float* query = (const float*)d_in[0];
    const float* key_  = (const float*)d_in[1];
    const float* value = (const float*)d_in[2];
    const int*   Mask  = (const int*)d_in[3];
    const float* Wq = (const float*)d_in[4];  const float* bq = (const float*)d_in[5];
    const float* Wk = (const float*)d_in[6];  const float* bk = (const float*)d_in[7];
    const float* Wv = (const float*)d_in[8];  const float* bv = (const float*)d_in[9];
    const float* Wm = (const float*)d_in[10]; const float* bm = (const float*)d_in[11];

    const int B = 4;
    unsigned short* Qb = (unsigned short*)d_ws;
    unsigned short* Kb = Qb + (size_t)B * 8 * SEQLEN * 64;
    unsigned short* Vb = Kb + (size_t)B * 8 * SEQLEN * 64;
    float*          Xf = (float*)(Vb + (size_t)B * 8 * SEQLEN * 64);

    dim3 blk(256);
    dim3 g1(SEQLEN / 64, DMODEL / 128, B);
    gemm_proj<0><<<g1, blk, 0, stream>>>(Wq, query, bq, Qb, nullptr);
    gemm_proj<0><<<g1, blk, 0, stream>>>(Wk, key_,  bk, Kb, nullptr);
    gemm_proj<0><<<g1, blk, 0, stream>>>(Wv, value, bv, Vb, nullptr);
    dim3 g2(SEQLEN / 128, 8, B);
    attn<<<g2, blk, 0, stream>>>(Qb, Kb, Vb, Mask, Xf);
    gemm_proj<1><<<g1, blk, 0, stream>>>(Wm, Xf, bm, nullptr, (float*)d_out);
}
